// P2RLossV2_63196148793592
// MI455X (gfx1250) — matrix operand order repaired and checked by reading, not compiled, not run on hardware
//
#include <hip/hip_runtime.h>
#include <hip/hip_bf16.h>
#include <math.h>

typedef __attribute__((ext_vector_type(16))) _Float16 v16h;
typedef __attribute__((ext_vector_type(8)))  _Float16 v8h;
typedef __attribute__((ext_vector_type(8)))  float    v8f;

#define EPSF 1e-8f
#define BATCH 16
#define NTOT 4096            // 64*64 OT grid
#define KT_TOTAL 128         // 4096 / 32 k-steps
#define NT_TOTAL 256         // 4096 / 16 column tiles
#define B_TILE_STRIDE (NT_TOTAL * 32 * 16)   // halves per kt slab = 131072

// ---------------------------------------------------------------------------
// K0: 1D gaussian weights, sigma=8, size 49, normalized (separable == 2D/2Dsum)
__global__ void k_gauss_weights(float* __restrict__ gw) {
    int t = threadIdx.x;
    if (t < 49) {
        float s = 0.f;
        for (int i = 0; i < 49; ++i) { float d = (float)(i - 24); s += __expf(-d * d / 128.f); }
        float d = (float)(t - 24);
        gw[t] = __expf(-d * d / 128.f) / s;
    }
}

// K1: 4x4 block sums: [16,512,512] -> [16,128,128] for pred and gt
__global__ void k_block4(const float* __restrict__ pred, const float* __restrict__ gt,
                         float* __restrict__ p4, float* __restrict__ g4) {
    int idx = blockIdx.x * blockDim.x + threadIdx.x;      // 16*128*128
    if (idx >= BATCH * 128 * 128) return;
    int c = idx & (128 * 128 - 1);
    int b = idx >> 14;
    int y = c >> 7, x = c & 127;
    size_t base = ((size_t)b << 18) + (size_t)(y * 4) * 512 + (size_t)(x * 4);
    float sp = 0.f, sg = 0.f;
    #pragma unroll
    for (int dy = 0; dy < 4; ++dy) {
        #pragma unroll
        for (int dx = 0; dx < 4; ++dx) {
            sp += pred[base + dy * 512 + dx];
            sg += gt[base + dy * 512 + dx];
        }
    }
    p4[idx] = sp;
    g4[idx] = sg;
}

// K2: 2x2 block sums of p4 -> a_raw/b_raw [16,4096], plus per-image totals
__global__ void k_block8_sums(const float* __restrict__ p4, const float* __restrict__ g4,
                              float* __restrict__ a, float* __restrict__ b,
                              float* __restrict__ a_sum, float* __restrict__ b_sum) {
    int img = blockIdx.x;
    int t = threadIdx.x;
    __shared__ float sa[256], sb[256];
    float la = 0.f, lb = 0.f;
    for (int i = t; i < NTOT; i += 256) {
        int y = i >> 6, x = i & 63;
        size_t o = (size_t)img * 16384 + (size_t)(y * 2) * 128 + (size_t)(x * 2);
        float va = p4[o] + p4[o + 1] + p4[o + 128] + p4[o + 129];
        float vb = g4[o] + g4[o + 1] + g4[o + 128] + g4[o + 129];
        a[(size_t)img * NTOT + i] = va;
        b[(size_t)img * NTOT + i] = vb;
        la += va; lb += vb;
    }
    sa[t] = la; sb[t] = lb;
    __syncthreads();
    for (int s = 128; s > 0; s >>= 1) {
        if (t < s) { sa[t] += sa[t + s]; sb[t] += sb[t + s]; }
        __syncthreads();
    }
    if (t == 0) { a_sum[img] = sa[0]; b_sum[img] = sb[0]; }
}

// K3: horizontal 49-tap conv, zero-padded, 32 images of 128x128 (p4|g4 contiguous)
__global__ void k_conv_h(const float* __restrict__ in, float* __restrict__ out,
                         const float* __restrict__ gw) {
    int row = blockIdx.x;           // 0 .. 32*128-1
    int x = threadIdx.x;            // 128
    __shared__ float srow[128];
    __shared__ float w[49];
    srow[x] = in[(size_t)row * 128 + x];
    if (x < 49) w[x] = gw[x];
    __syncthreads();
    float acc = 0.f;
    #pragma unroll
    for (int d = -24; d <= 24; ++d) {
        int xx = x + d;
        if (xx >= 0 && xx < 128) acc += w[d + 24] * srow[xx];
    }
    out[(size_t)row * 128 + x] = acc;
}

// K4: vertical 49-tap conv
__global__ void k_conv_v(const float* __restrict__ in, float* __restrict__ out,
                         const float* __restrict__ gw) {
    int col = blockIdx.x & 127;
    int img = blockIdx.x >> 7;      // 0..31
    int y = threadIdx.x;            // 128
    __shared__ float scol[128];
    __shared__ float w[49];
    scol[y] = in[(size_t)img * 16384 + (size_t)y * 128 + col];
    if (y < 49) w[y] = gw[y];
    __syncthreads();
    float acc = 0.f;
    #pragma unroll
    for (int d = -24; d <= 24; ++d) {
        int yy = y + d;
        if (yy >= 0 && yy < 128) acc += w[d + 24] * scol[yy];
    }
    out[(size_t)img * 16384 + (size_t)y * 128 + col] = acc;
}

// K5: per-image sums of smoothed densities
__global__ void k_img_sum2(const float* __restrict__ ps, const float* __restrict__ gs,
                           float* __restrict__ s_ps, float* __restrict__ s_gs) {
    int img = blockIdx.x;
    int t = threadIdx.x;
    __shared__ float sa[256], sb[256];
    float la = 0.f, lb = 0.f;
    for (int i = t; i < 16384; i += 256) {
        la += ps[(size_t)img * 16384 + i];
        lb += gs[(size_t)img * 16384 + i];
    }
    sa[t] = la; sb[t] = lb;
    __syncthreads();
    for (int s = 128; s > 0; s >>= 1) {
        if (t < s) { sa[t] += sa[t + s]; sb[t] += sb[t + s]; }
        __syncthreads();
    }
    if (t == 0) { s_ps[img] = sa[0]; s_gs[img] = sb[0]; }
}

// K6: per-image KL + L2 on normalized smoothed maps
__global__ void k_kl_l2(const float* __restrict__ ps, const float* __restrict__ gs,
                        const float* __restrict__ s_ps, const float* __restrict__ s_gs,
                        float* __restrict__ kl, float* __restrict__ l2) {
    int img = blockIdx.x;
    int t = threadIdx.x;
    __shared__ float sk[256], sl[256];
    float inv_p = 1.f / fmaxf(s_ps[img], EPSF);
    float inv_g = 1.f / fmaxf(s_gs[img], EPSF);
    float lk = 0.f, ll = 0.f;
    for (int i = t; i < 16384; i += 256) {
        float pn = ps[(size_t)img * 16384 + i] * inv_p;
        float gn = gs[(size_t)img * 16384 + i] * inv_g;
        lk += gn * __logf((gn + EPSF) / (pn + EPSF));
        float d = pn - gn;
        ll += d * d;
    }
    sk[t] = lk; sl[t] = ll;
    __syncthreads();
    for (int s = 128; s > 0; s >>= 1) {
        if (t < s) { sk[t] += sk[t + s]; sl[t] += sl[t + s]; }
        __syncthreads();
    }
    if (t == 0) { kl[img] = sk[0]; l2[img] = sl[0]; }
}

// K7: build K = exp(-10*C) and K*C in f16, pre-swizzled into the WMMA B layout.
// Packed index = ((kt*256 + nt)*32 + lane)*16 + e,
// with k = kt*32 + (lane>>4)*8 + (e&7) + (e>>3)*16,  n = nt*16 + (lane&15).
__global__ void k_build_K(_Float16* __restrict__ Kp, _Float16* __restrict__ KCp) {
    unsigned idx = blockIdx.x * 256u + threadIdx.x;       // 16,777,216 total
    int e    = idx & 15;
    int lane = (idx >> 4) & 31;
    int nt   = (idx >> 9) & 255;
    int kt   = idx >> 17;
    int k = kt * 32 + ((lane >> 4) * 8) + (e & 7) + ((e >> 3) * 16);
    int n = nt * 16 + (lane & 15);
    float dy = (float)((k >> 6) - (n >> 6));
    float dx = (float)((k & 63) - (n & 63));
    float C = dy * dy + dx * dx;
    float Kv = __expf(-10.f * C);                          // exp(-C/OT_REG), OT_REG=0.1
    Kp[idx]  = (_Float16)Kv;
    KCp[idx] = (_Float16)(Kv * C);
}

// K8: normalize a,b in place; init u=v=1 (f32 + f16 operands)
__global__ void k_norm_init(float* __restrict__ a, float* __restrict__ b,
                            const float* __restrict__ a_sum, const float* __restrict__ b_sum,
                            float* __restrict__ u, float* __restrict__ v,
                            _Float16* __restrict__ u_h, _Float16* __restrict__ v_h) {
    int idx = blockIdx.x * 256 + threadIdx.x;             // 16*4096
    int img = idx >> 12;
    a[idx] = a[idx] / fmaxf(a_sum[img], EPSF);
    b[idx] = b[idx] / fmaxf(b_sum[img], EPSF);
    u[idx] = 1.f; v[idx] = 1.f;
    u_h[idx] = (_Float16)1.f; v_h[idx] = (_Float16)1.f;
}

// K9: WMMA Sinkhorn GEMM: acc[16,nt*16..] = A[16,4096] @ Bpacked, 4-wave split-K,
// manually unrolled x4 so 16 b128 loads are in flight before 4 back-to-back WMMAs.
// mode 0: dst = numer/(acc+eps) (f32 + f16); mode 1: dst = acc (f32 only).
__global__ void __launch_bounds__(128)
k_sink_gemm(const _Float16* __restrict__ A, const _Float16* __restrict__ Bp,
            const float* __restrict__ numer, float* __restrict__ dst,
            _Float16* __restrict__ dst_h, int mode) {
    int nt   = blockIdx.x;              // 0..255 column tile
    int lane = threadIdx.x & 31;
    int wave = threadIdx.x >> 5;        // 0..3, split over kt
    int m    = lane & 15;               // A-operand row for this lane
    int koff = (lane >> 4) * 8;         // A-operand k-half select

    v8f acc = {};
    const _Float16* paBase = A + (size_t)m * NTOT + koff;
    const _Float16* pbBase = Bp + ((size_t)nt * 32 + lane) * 16;

    int kt0 = wave * 32;
    for (int kt = kt0; kt < kt0 + 32; kt += 4) {
        v16h av[4], bv[4];
        #pragma unroll
        for (int j = 0; j < 4; ++j) {
            const _Float16* pa = paBase + (kt + j) * 32;
            v8h lo = *(const v8h*)(pa);          // k = k0+koff .. +7
            v8h hi = *(const v8h*)(pa + 16);     // k = k0+koff+16 .. +23
            #pragma unroll
            for (int i = 0; i < 8; ++i) { av[j][i] = lo[i]; av[j][i + 8] = hi[i]; }
            bv[j] = *(const v16h*)(pbBase + (size_t)(kt + j) * B_TILE_STRIDE);
        }
        #pragma unroll
        for (int j = 0; j < 4; ++j)
            acc = __builtin_amdgcn_wmma_f32_16x16x32_f16(
                false, av[j], false, bv[j], (short)0, acc, false, false);
    }

    __shared__ float red[4 * 32 * 8];
    #pragma unroll
    for (int r = 0; r < 8; ++r) red[(wave * 32 + lane) * 8 + r] = acc[r];
    __syncthreads();

    if (wave == 0) {
        float tot[8];
        #pragma unroll
        for (int r = 0; r < 8; ++r)
            tot[r] = red[lane * 8 + r] + red[(32 + lane) * 8 + r] +
                     red[(64 + lane) * 8 + r] + red[(96 + lane) * 8 + r];
        // f32 C/D layout: VGPR r, lanes 0-15 -> M=r; lanes 16-31 -> M=r+8; N=lane&15
        int n = nt * 16 + (lane & 15);
        int mbase = (lane >> 4) * 8;
        #pragma unroll
        for (int r = 0; r < 8; ++r) {
            size_t o = (size_t)(mbase + r) * NTOT + n;
            if (mode == 0) {
                float val = numer[o] / (tot[r] + EPSF);
                dst[o] = val;
                dst_h[o] = (_Float16)val;
            } else {
                dst[o] = tot[r];
            }
        }
    }
}

// K11: cost[b] = sum_n u[b,n] * t[b,n]
__global__ void k_ot_cost(const float* __restrict__ u, const float* __restrict__ t,
                          float* __restrict__ cost) {
    int img = blockIdx.x;
    int tid = threadIdx.x;
    __shared__ float s[256];
    float l = 0.f;
    for (int i = tid; i < NTOT; i += 256)
        l += u[(size_t)img * NTOT + i] * t[(size_t)img * NTOT + i];
    s[tid] = l;
    __syncthreads();
    for (int st = 128; st > 0; st >>= 1) {
        if (tid < st) s[tid] += s[tid + st];
        __syncthreads();
    }
    if (tid == 0) cost[img] = s[0];
}

// K12: final scalar combination
__global__ void k_combine(const float* __restrict__ a_sum, const float* __restrict__ b_sum,
                          const int* __restrict__ gt_counts, const int* __restrict__ cell_area,
                          const float* __restrict__ kl, const float* __restrict__ l2,
                          const float* __restrict__ cost, float* __restrict__ out) {
    if (threadIdx.x != 0 || blockIdx.x != 0) return;
    float ca = (float)cell_area[0];
    float cl = 0.f, dmkl = 0.f, dml2 = 0.f, ot = 0.f;
    for (int b = 0; b < BATCH; ++b) {
        float pc = fmaxf(a_sum[b] / ca, 0.f);
        float diff = fabsf(pc - (float)gt_counts[b]);
        cl += (diff < 10.f) ? (0.5f * diff * diff / 10.f) : (diff - 5.f);
        dmkl += kl[b];
        dml2 += l2[b];
        bool mask = (a_sum[b] > 0.5f) && (b_sum[b] > 0.5f);
        ot += mask ? cost[b] : 0.f;
    }
    cl /= (float)BATCH;
    float dm = dmkl / (float)BATCH + dml2 / (float)BATCH;
    ot /= (float)BATCH;
    out[0] = 3.0f * cl + 0.5f * dm + 0.3f * ot;
}

// ---------------------------------------------------------------------------
extern "C" void kernel_launch(void* const* d_in, const int* in_sizes, int n_in,
                              void* d_out, int out_size, void* d_ws, size_t ws_size,
                              hipStream_t stream) {
    const float* pred = (const float*)d_in[0];
    const float* gt   = (const float*)d_in[1];
    const int*   gtc  = (const int*)d_in[2];
    const int*   ca   = (const int*)d_in[3];

    char* ws = (char*)d_ws;
    size_t off = 0;
    auto take = [&](size_t bytes) -> char* {
        char* p = ws + off;
        off += (bytes + 255) & ~(size_t)255;
        return p;
    };

    _Float16* Kp   = (_Float16*)take((size_t)NTOT * NTOT * 2);   // 32 MB, L2-resident
    _Float16* KCp  = (_Float16*)take((size_t)NTOT * NTOT * 2);   // 32 MB
    float* pg4  = (float*)take((size_t)2 * BATCH * 128 * 128 * 4);  // p4|g4 contiguous
    float* tmp  = (float*)take((size_t)2 * BATCH * 128 * 128 * 4);  // conv intermediate
    float* psgs = (float*)take((size_t)2 * BATCH * 128 * 128 * 4);  // ps|gs contiguous
    float* a    = (float*)take((size_t)BATCH * NTOT * 4);
    float* b    = (float*)take((size_t)BATCH * NTOT * 4);
    float* u    = (float*)take((size_t)BATCH * NTOT * 4);
    float* v    = (float*)take((size_t)BATCH * NTOT * 4);
    float* t    = (float*)take((size_t)BATCH * NTOT * 4);
    _Float16* u_h = (_Float16*)take((size_t)BATCH * NTOT * 2);
    _Float16* v_h = (_Float16*)take((size_t)BATCH * NTOT * 2);
    float* scal = (float*)take(512 * 4);
    float* a_sum = scal;       float* b_sum = scal + 16;
    float* s_ps  = scal + 32;  float* s_gs  = scal + 48;
    float* kl    = scal + 64;  float* l2    = scal + 80;
    float* cost  = scal + 96;  float* gw    = scal + 128;   // 49 weights

    float* p4 = pg4;  float* g4 = pg4 + (size_t)BATCH * 128 * 128;
    float* ps = psgs; float* gs = psgs + (size_t)BATCH * 128 * 128;

    // --- count prep + DM path ---
    k_gauss_weights<<<1, 64, 0, stream>>>(gw);
    k_block4<<<(BATCH * 128 * 128 + 255) / 256, 256, 0, stream>>>(pred, gt, p4, g4);
    k_block8_sums<<<BATCH, 256, 0, stream>>>(p4, g4, a, b, a_sum, b_sum);
    k_conv_h<<<32 * 128, 128, 0, stream>>>(pg4, tmp, gw);
    k_conv_v<<<32 * 128, 128, 0, stream>>>(tmp, psgs, gw);
    k_img_sum2<<<BATCH, 256, 0, stream>>>(ps, gs, s_ps, s_gs);
    k_kl_l2<<<BATCH, 256, 0, stream>>>(ps, gs, s_ps, s_gs, kl, l2);

    // --- OT path: build packed kernels, init, 50 Sinkhorn iterations ---
    k_build_K<<<(NTOT * NTOT) / 256, 256, 0, stream>>>(Kp, KCp);
    k_norm_init<<<(BATCH * NTOT) / 256, 256, 0, stream>>>(a, b, a_sum, b_sum, u, v, u_h, v_h);

    for (int it = 0; it < 50; ++it) {
        // v = b / (u @ K + eps)      (K symmetric)
        k_sink_gemm<<<NT_TOTAL, 128, 0, stream>>>(u_h, Kp, b, v, v_h, 0);
        // u = a / (v @ K + eps)
        k_sink_gemm<<<NT_TOTAL, 128, 0, stream>>>(v_h, Kp, a, u, u_h, 0);
    }
    // t = v @ (K*C)   (symmetric), then cost = <u, t>
    k_sink_gemm<<<NT_TOTAL, 128, 0, stream>>>(v_h, KCp, a, t, v_h, 1);
    k_ot_cost<<<BATCH, 256, 0, stream>>>(u, t, cost);

    k_combine<<<1, 32, 0, stream>>>(a_sum, b_sum, gtc, ca, kl, l2, cost, (float*)d_out);
}
